// RVQTokenizer_1580547965071
// MI455X (gfx1250) — matrix operand order repaired and checked by reading, not compiled, not run on hardware
//
#include <hip/hip_runtime.h>
#include <hip/hip_bf16.h>
#include <stdint.h>

// ---------------------------------------------------------------------------
// Residual VQ forward for MI455X (gfx1250).
//   B=16, T=2048, D=256, Q=4, K=1024.  GEMM per stage: 32768x256 * 256x1024.
// Strategy: bf16x3 split-precision WMMA (v_wmma_f32_16x16x32_bf16) so the
// matrix engine runs the cdist GEMM at ~fp32 accuracy while staying at/under
// the 23.3 TB/s HBM roofline (~9us of unavoidable traffic).
// ---------------------------------------------------------------------------

typedef __attribute__((ext_vector_type(16))) __bf16 v16bf;
typedef __attribute__((ext_vector_type(8)))  float  v8f;

union Frag { v16bf bf; uint32_t u[8]; };
union Acc8 { v8f v; float f[8]; };

__device__ __forceinline__ uint16_t f2bf(float f) {
  union { float f; uint32_t u; } v; v.f = f;
  uint32_t r = v.u + 0x7FFFu + ((v.u >> 16) & 1u);   // round-to-nearest-even
  return (uint16_t)(r >> 16);
}
__device__ __forceinline__ float bf2f(uint16_t h) {
  union { uint32_t u; float f; } v; v.u = ((uint32_t)h) << 16;
  return v.f;
}

#define NBT   32768        // B*T tokens
#define DD    256
#define KK    1024
#define QQ    4
#define TM    64           // tokens per block
#define CHUNK 64           // codes per LDS chunk
#define NCH   (KK / CHUNK) // 16
#define AP    264          // padded LDS row stride in halves (bank spread)
#define DP    66           // padded dist row stride in floats

// ---------------------------------------------------------------------------
// Prep: split all codebooks into bf16 hi/lo planes, compute ||c||^2, zero
// the loss partial buffer (re-done every call -> graph-replay safe).
// grid = Q*K blocks of 256 threads (one code row per block).
// ---------------------------------------------------------------------------
__global__ void rvq_prep(const float* __restrict__ cb,
                         uint16_t* __restrict__ cbhi, uint16_t* __restrict__ cblo,
                         float* __restrict__ cnorm, float* __restrict__ partial) {
  const int row = blockIdx.x;        // 0 .. Q*K-1
  const int t   = threadIdx.x;       // 0 .. 255  (== D)
  float c = cb[row * DD + t];
  uint16_t hi = f2bf(c);
  cbhi[row * DD + t] = hi;
  cblo[row * DD + t] = f2bf(c - bf2f(hi));
  __shared__ float red[256];
  red[t] = c * c;
  __syncthreads();
  for (int s = 128; s > 0; s >>= 1) { if (t < s) red[t] += red[t + s]; __syncthreads(); }
  if (t == 0) cnorm[row] = red[0];
  if (row < QQ * (NBT / TM) && t == 0) partial[row] = 0.0f;  // defensive zero
}

// ---------------------------------------------------------------------------
// One RVQ stage: distances via WMMA, argmin, gather/update/loss partial.
// grid = NBT/TM = 512 blocks of 256 threads (8 waves).
// Waves: mt = w/4 owns rows [32*mt, 32*mt+32) as two 16-row tiles,
//        nt = w%4 owns 16 codes of the 64-code chunk.
// ---------------------------------------------------------------------------
__global__ __launch_bounds__(256)
void rvq_stage(const float* __restrict__ resid_in, float* __restrict__ resid_out,
               const float* __restrict__ cb32,
               const uint16_t* __restrict__ cbhi, const uint16_t* __restrict__ cblo,
               const float* __restrict__ cnorm,
               float* __restrict__ quant, int* __restrict__ idx_out,
               float* __restrict__ partial, int first) {
  __shared__ uint16_t a_hi[TM * AP];         // residual hi plane
  __shared__ uint16_t a_lo[TM * AP];         // residual lo plane
  __shared__ uint16_t b_hi[CHUNK * AP];      // codebook chunk hi
  __shared__ uint16_t b_lo[CHUNK * AP];      // codebook chunk lo
  __shared__ float    dist[TM * DP];
  __shared__ float    pval[TM * 4];
  __shared__ int      pidx[TM * 4];
  __shared__ float    best_d[TM];
  __shared__ int      best_i[TM];
  __shared__ float    red[256];

  const int t    = threadIdx.x;
  const int wave = t >> 5;
  const int lane = t & 31;
  const int tok0 = (int)blockIdx.x * TM;

  // ---- load + split the residual tile (64 rows x 256 dims) ----
  for (int i = t; i < TM * DD; i += 256) {
    const int r = i >> 8, d = i & 255;
    const float x = resid_in[(size_t)(tok0 + r) * DD + d];
    const uint16_t hi = f2bf(x);
    a_hi[r * AP + d] = hi;
    a_lo[r * AP + d] = f2bf(x - bf2f(hi));
  }
  if (t < TM) { best_d[t] = 3.4e38f; best_i[t] = 0; }
  __syncthreads();

  const int mt = wave >> 2;                 // 0..1
  const int nt = wave & 3;                  // 0..3
  const int hh = lane >> 4;                 // half-wave select
  const int arow0 = mt * 32 + (lane & 15);  // first 16-row tile
  const int arow1 = arow0 + 16;             // second 16-row tile
  const int brow  = nt * 16 + (lane & 15);  // code row within chunk

  const uint32_t* ahp = (const uint32_t*)a_hi;
  const uint32_t* alp = (const uint32_t*)a_lo;
  const uint32_t* bhp = (const uint32_t*)b_hi;
  const uint32_t* blp = (const uint32_t*)b_lo;

  for (int ch = 0; ch < NCH; ++ch) {
    // ---- cooperative load of codebook chunk (both planes) as dwords ----
    {
      const uint32_t* shi = (const uint32_t*)(cbhi + (size_t)(ch * CHUNK) * DD);
      const uint32_t* slo = (const uint32_t*)(cblo + (size_t)(ch * CHUNK) * DD);
      uint32_t* dhi = (uint32_t*)b_hi;
      uint32_t* dlo = (uint32_t*)b_lo;
      for (int i = t; i < CHUNK * (DD / 2); i += 256) {
        const int r = i >> 7, o = i & 127;
        dhi[r * (AP / 2) + o] = shi[i];
        dlo[r * (AP / 2) + o] = slo[i];
      }
      if (ch + 1 < NCH) {  // gfx1250 global_prefetch_b8 on next chunk
        __builtin_prefetch(cbhi + (size_t)((ch + 1) * CHUNK) * DD + t * 64, 0, 1);
        __builtin_prefetch(cblo + (size_t)((ch + 1) * CHUNK) * DD + t * 64, 0, 1);
      }
    }
    __syncthreads();

    // ---- GEMM: two 16x16 output tiles per wave, bf16x3 accumulation ----
    Acc8 c0, c1;
#pragma unroll
    for (int v = 0; v < 8; ++v) { c0.f[v] = 0.0f; c1.f[v] = 0.0f; }

#pragma unroll
    for (int kb = 0; kb < DD / 32; ++kb) {
      const int kbase = kb * 32;
      Frag ah0, al0, ah1, al1, bh, bl;
      // B fragment (KxN): lane holds N=lane%16; dword v<4 -> k=kbase+8*hh+2v ; v>=4 -> +16
      const int bo = brow * (AP / 2) + (kbase + 8 * hh) / 2;
#pragma unroll
      for (int v = 0; v < 8; ++v) { bh.u[v] = bhp[bo + v + (v >= 4 ? 4 : 0)]; bl.u[v] = blp[bo + v + (v >= 4 ? 4 : 0)]; }
      // A fragment (MxK): dword v<4 -> k=kbase+8*hh+2v ; v>=4 -> +16
      const int ao0 = arow0 * (AP / 2) + (kbase + 8 * hh) / 2;
      const int ao1 = arow1 * (AP / 2) + (kbase + 8 * hh) / 2;
#pragma unroll
      for (int v = 0; v < 4; ++v) {
        ah0.u[v] = ahp[ao0 + v];  ah0.u[v + 4] = ahp[ao0 + 8 + v];
        al0.u[v] = alp[ao0 + v];  al0.u[v + 4] = alp[ao0 + 8 + v];
        ah1.u[v] = ahp[ao1 + v];  ah1.u[v + 4] = ahp[ao1 + 8 + v];
        al1.u[v] = alp[ao1 + v];  al1.u[v + 4] = alp[ao1 + 8 + v];
      }
      c0.v = __builtin_amdgcn_wmma_f32_16x16x32_bf16(false, ah0.bf, false, bh.bf, (short)0, c0.v, false, false);
      c0.v = __builtin_amdgcn_wmma_f32_16x16x32_bf16(false, ah0.bf, false, bl.bf, (short)0, c0.v, false, false);
      c0.v = __builtin_amdgcn_wmma_f32_16x16x32_bf16(false, al0.bf, false, bh.bf, (short)0, c0.v, false, false);
      c1.v = __builtin_amdgcn_wmma_f32_16x16x32_bf16(false, ah1.bf, false, bh.bf, (short)0, c1.v, false, false);
      c1.v = __builtin_amdgcn_wmma_f32_16x16x32_bf16(false, ah1.bf, false, bl.bf, (short)0, c1.v, false, false);
      c1.v = __builtin_amdgcn_wmma_f32_16x16x32_bf16(false, al1.bf, false, bh.bf, (short)0, c1.v, false, false);
    }

    // ---- distances: ||c||^2 - 2*(r.c)  (||r||^2 is argmin-invariant) ----
    const int ncol = nt * 16 + (lane & 15);
    const float cn = cnorm[ch * CHUNK + ncol];
#pragma unroll
    for (int v = 0; v < 8; ++v) {     // C layout: lane n=lane%16, row m=8*hh+v
      const int m0 = mt * 32 + 8 * hh + v;
      dist[m0 * DP + ncol]        = cn - 2.0f * c0.f[v];
      dist[(m0 + 16) * DP + ncol] = cn - 2.0f * c1.f[v];
    }
    __syncthreads();

    // ---- argmin phase 1: 4 segments x 16 cols per row ----
    {
      const int r = t >> 2, s = t & 3, cb0 = s * 16;
      float bv = 3.4e38f; int bi = cb0;
      for (int c = 0; c < 16; ++c) {
        const float dv = dist[r * DP + cb0 + c];
        if (dv < bv) { bv = dv; bi = cb0 + c; }       // strict < : first index
      }
      pval[r * 4 + s] = bv; pidx[r * 4 + s] = ch * CHUNK + bi;
    }
    __syncthreads();
    // ---- argmin phase 2: fold segments into running best (ascending n) ----
    if (t < TM) {
      float bv = best_d[t]; int bi = best_i[t];
      for (int s = 0; s < 4; ++s) {
        const float v = pval[t * 4 + s];
        if (v < bv) { bv = v; bi = pidx[t * 4 + s]; }
      }
      best_d[t] = bv; best_i[t] = bi;
    }
    __syncthreads();
  }

  // ---- gather, residual update, quant accumulate, loss partial ----
  // loss/stage = 1.25 * mean(residual_new^2)  (codebook_loss == commit_loss)
  float ssq = 0.0f;
  for (int i = 0; i < TM; ++i) {
    const size_t e = (size_t)(tok0 + i) * DD + t;   // dim = t -> coalesced
    const float qv   = cb32[(size_t)best_i[i] * DD + t];
    const float rn   = resid_in[e] - qv;            // read-before-write: in-place OK
    resid_out[e] = rn;
    quant[e] = first ? qv : (quant[e] + qv);
    ssq += rn * rn;
  }
  if (t < TM) idx_out[tok0 + t] = best_i[t];
  red[t] = ssq;
  __syncthreads();
  for (int s = 128; s > 0; s >>= 1) { if (t < s) red[t] += red[t + s]; __syncthreads(); }
  if (t == 0) partial[blockIdx.x] = red[0];
}

// ---------------------------------------------------------------------------
// Finalize: fixed-order (deterministic) sum of loss partials -> 4 scalars.
// ---------------------------------------------------------------------------
__global__ void rvq_finalize(const float* __restrict__ partial, float* __restrict__ loss_out) {
  __shared__ float red[256];
  __shared__ float qs[QQ];
  const int t = threadIdx.x;
  for (int q = 0; q < QQ; ++q) {
    float s = 0.0f;
    for (int i = t; i < NBT / TM; i += 256) s += partial[q * (NBT / TM) + i];
    red[t] = s; __syncthreads();
    for (int k = 128; k > 0; k >>= 1) { if (t < k) red[t] += red[t + k]; __syncthreads(); }
    if (t == 0) qs[q] = red[0];
    __syncthreads();
  }
  float total = 0.0f;
  const float invN = 1.0f / (float)((size_t)NBT * DD);
  for (int q = 0; q < QQ; ++q) total += 1.25f * qs[q] * invN;
  if (t < QQ) loss_out[t] = total;
}

// ---------------------------------------------------------------------------
extern "C" void kernel_launch(void* const* d_in, const int* in_sizes, int n_in,
                              void* d_out, int out_size, void* d_ws, size_t ws_size,
                              hipStream_t stream) {
  const float* x  = (const float*)d_in[0];   // [B,T,D] f32
  const float* cb = (const float*)d_in[1];   // [Q,K,D] f32
  float* out = (float*)d_out;

  const size_t BTD = (size_t)NBT * DD;       // 8,388,608

  // workspace layout
  char* ws = (char*)d_ws;
  float*    resid   = (float*)ws;      ws += BTD * sizeof(float);            // 33.5 MB
  uint16_t* cbhi    = (uint16_t*)ws;   ws += (size_t)QQ * KK * DD * 2;       // 2 MB
  uint16_t* cblo    = (uint16_t*)ws;   ws += (size_t)QQ * KK * DD * 2;       // 2 MB
  float*    cnorm   = (float*)ws;      ws += (size_t)QQ * KK * sizeof(float);
  float*    partial = (float*)ws;      ws += (size_t)QQ * (NBT / TM) * sizeof(float);

  // output layout: quant [B,T,D] f32 | indices [Q,B,T] i32 | loss x4
  float* quant    = out;
  int*   idx_base = (int*)(out + BTD);
  float* loss_out = out + BTD + (size_t)QQ * NBT;

  rvq_prep<<<QQ * KK, 256, 0, stream>>>(cb, cbhi, cblo, cnorm, partial);

  for (int q = 0; q < QQ; ++q) {
    rvq_stage<<<NBT / TM, 256, 0, stream>>>(
        q == 0 ? x : resid, resid,
        cb   + (size_t)q * KK * DD,
        cbhi + (size_t)q * KK * DD,
        cblo + (size_t)q * KK * DD,
        cnorm + (size_t)q * KK,
        quant, idx_base + (size_t)q * NBT,
        partial + (size_t)q * (NBT / TM),
        q == 0 ? 1 : 0);
  }

  rvq_finalize<<<1, 256, 0, stream>>>(partial, loss_out);
}
